// CrossAttention_56977036149109
// MI455X (gfx1250) — matrix operand order repaired and checked by reading, not compiled, not run on hardware
//
#include <hip/hip_runtime.h>
#include <hip/hip_bf16.h>

// ---------------- problem constants ----------------
#define B_     2
#define NV     6
#define DIMC   128
#define HWQ    1024        // 32*32 query tokens per (b,n)
#define HWK    1680        // 28*60 key tokens per (b,n)
#define NKTOT  (NV*HWK)    // 10080 joint keys
#define HEADS_ 4
#define DH_    32
#define INNER_ 128

typedef _Float16 half_t;
typedef __attribute__((ext_vector_type(16))) _Float16 v16h;
typedef __attribute__((ext_vector_type(8)))  _Float16 v8h;
typedef __attribute__((ext_vector_type(8)))  float    v8f;

#define WMMA_F16(a,b,c) __builtin_amdgcn_wmma_f32_16x16x32_f16(false,(a),false,(b),(short)0,(c),false,false)
#define LDS_FENCE() asm volatile("s_wait_dscnt 0x0" ::: "memory")

// A-frag (16x32 f16, M=row in lane, K across VGPRs). Caller passes this lane's
// row pointer; k0/k1 are the two 8-half K windows for this lane half.
__device__ __forceinline__ v16h afrag_load(const half_t* row, int k0, int k1) {
  v8h lo = *(const v8h*)(row + k0);
  v8h hi = *(const v8h*)(row + k1);
  v16h a;
#pragma unroll
  for (int i = 0; i < 8; ++i) { a[i] = lo[i]; a[8 + i] = hi[i]; }
  return a;
}

// B-frag (32x16 f16) from 16 contiguous halves (lane's K window, N=lane%16 row).
__device__ __forceinline__ v16h bfrag_load16(const half_t* p) {
  v8h lo = *(const v8h*)(p);
  v8h hi = *(const v8h*)(p + 8);
  v16h b;
#pragma unroll
  for (int i = 0; i < 8; ++i) { b[i] = lo[i]; b[8 + i] = hi[i]; }
  return b;
}

// B-frag from pre-converted f16 weight, row-major (out,in): B[k][o] = W[o][k].
// kb already includes the +16 for high lanes.
__device__ __forceinline__ v16h bfrag_wh(const half_t* W, int ld, int o, int kb) {
  return bfrag_load16(W + (size_t)o * ld + kb);
}

__device__ __forceinline__ unsigned pk2(float a, float b) {
  union { _Float16 h[2]; unsigned u; } x;
  x.h[0] = (_Float16)a; x.h[1] = (_Float16)b;
  return x.u;
}

// ================= kernel 0: one-time f32 -> f16 weight convert =================
__global__ __launch_bounds__(256) void cvt_kernel(const float* __restrict__ src,
                                                  half_t* __restrict__ dst, int n) {
  int i = blockIdx.x * 256 + threadIdx.x;
  if (i < n) dst[i] = (_Float16)src[i];
}

// ================= kernel 1: LN + linear projection =================
// x: (B, NV, DIMC, HW) f32.  out (transposed==0): (B*NV*HW, 128) f16 row-major.
// out (transposed==1): (B, 128, NKTOT) f16 (value matrix pre-transposed).
__global__ __launch_bounds__(128) void proj_kernel(
    const float* __restrict__ x, const float* __restrict__ lng, const float* __restrict__ lnb,
    const half_t* __restrict__ Wh, const float* __restrict__ bias,
    half_t* __restrict__ out, int HW, float scale, int transposed)
{
  __shared__ alignas(16) float  raw_s[4][16 * 128];
  __shared__ alignas(16) half_t at_s[4][16 * 128];
  int w = threadIdx.x >> 5;
  int lane = threadIdx.x & 31, l16 = lane & 15, hi = lane >> 4;
  int tile = blockIdx.x * 4 + w;
  int tb = tile * 16;
  if (tb >= B_ * NV * HW) return;
  int b = tb / (NV * HW); int rem = tb - b * (NV * HW);
  int vn = rem / HW;      int sp = rem - vn * HW;
  float*  raw = raw_s[w];
  half_t* at  = at_s[w];

  // gather + LN stats: lane handles token l16, channel half [hi*64, hi*64+64)
  const float* xin = x + ((size_t)(b * NV + vn) * DIMC) * HW + sp + l16;
  float s1 = 0.f, s2 = 0.f;
  int cbase = hi * 64;
#pragma unroll 8
  for (int c = 0; c < 64; ++c) {
    float v = xin[(size_t)(cbase + c) * HW];
    raw[l16 * 128 + cbase + c] = v;
    s1 += v; s2 += v * v;
  }
  s1 += __shfl_xor(s1, 16);
  s2 += __shfl_xor(s2, 16);
  float mean = s1 * (1.f / 128.f);
  float rstd = rsqrtf(s2 * (1.f / 128.f) - mean * mean + 1e-5f);
#pragma unroll 8
  for (int c = 0; c < 64; ++c) {
    int cc = cbase + c;
    float v = (raw[l16 * 128 + cc] - mean) * rstd * lng[cc] + lnb[cc];
    at[l16 * 128 + cc] = (_Float16)v;
  }
  LDS_FENCE();

  // (16 tokens x 128) @ W^T : 8 N-tiles x 4 K-chunks of WMMA
#pragma unroll
  for (int nt = 0; nt < 8; ++nt) {
    int o = nt * 16 + l16;
    v8f acc;
#pragma unroll
    for (int r = 0; r < 8; ++r) acc[r] = bias[o];
#pragma unroll
    for (int kc = 0; kc < 4; ++kc) {
      v16h a  = afrag_load(at + l16 * 128, kc * 32 + (hi ? 16 : 0), kc * 32 + (hi ? 24 : 8));
      v16h bf = bfrag_wh(Wh, 128, o, kc * 32 + hi * 16);
      acc = WMMA_F16(a, bf, acc);
    }
#pragma unroll
    for (int r = 0; r < 8; ++r) acc[r] *= scale;
    if (!transposed) {
      half_t* orow = out + (size_t)tb * INNER_;
#pragma unroll
      for (int r = 0; r < 8; ++r)
        orow[(size_t)(r + hi * 8) * INNER_ + o] = (_Float16)acc[r];
    } else {  // V^T: row = out-channel o, cols = key tokens (contiguous)
      half_t* orow = out + ((size_t)b * INNER_ + o) * NKTOT + vn * HWK + sp + hi * 8;
#pragma unroll
      for (int j = 0; j < 4; ++j)
        *(unsigned*)(orow + 2 * j) = pk2(acc[2 * j], acc[2 * j + 1]);
    }
  }
}

// ================= kernel 2: flash cross-attention (S^T form) =================
__global__ __launch_bounds__(128) void attn_kernel(
    const half_t* __restrict__ qproj, const half_t* __restrict__ kproj,
    const half_t* __restrict__ vt, half_t* __restrict__ attout)
{
  int w = threadIdx.x >> 5;
  int lane = threadIdx.x & 31, l16 = lane & 15, hi = lane >> 4;
  int wave = blockIdx.x * 4 + w;           // 512 waves: (b, head, qtile)
  int qt = wave & 63, hd = (wave >> 6) & 3, b = wave >> 8;
  int T0 = qt * 16;

  v8f O0{}, O1{};                          // O^T accumulators: dh 0-15 / 16-31 x 16 queries
  float m_run = -3.0e38f, l_run = 0.f;
  int ka0 = hd * 32 + (hi ? 16 : 0), ka1 = hd * 32 + (hi ? 24 : 8);
  const half_t* v0row = vt + ((size_t)(b * INNER_) + hd * 32 + l16) * NKTOT;
  const half_t* v1row = v0row + (size_t)16 * NKTOT;
  const half_t* qbase = qproj + ((size_t)(b * NV) * HWQ + T0 + l16) * INNER_ + hd * 32 + hi * 16;

  int nv_cur = -1;
  v16h bq{};
  for (int cp = 0; cp < NKTOT / 32; ++cp) {
    int g0 = cp * 32;
    v8f S[2];
#pragma unroll
    for (int cc = 0; cc < 2; ++cc) {
      int gb = g0 + cc * 16;
      int nv = gb / HWK;                   // view index of this key chunk
      if (nv != nv_cur) {                  // Q fragment changes per view only
        nv_cur = nv;
        bq = bfrag_load16(qbase + (size_t)nv * HWQ * INNER_);
      }
      const half_t* krow = kproj + ((size_t)b * NKTOT + gb + l16) * INNER_;
      v16h a = afrag_load(krow, ka0, ka1);
      v8f z{};
      S[cc] = WMMA_F16(a, bq, z);          // S^T tile: rows=keys, cols=queries
    }
    // online softmax over keys (rows): own 16 rows + partner lane (xor 16)
    float mx = -3.0e38f;
#pragma unroll
    for (int r = 0; r < 8; ++r) { mx = fmaxf(mx, S[0][r]); mx = fmaxf(mx, S[1][r]); }
    mx = fmaxf(mx, __shfl_xor(mx, 16));
    float m_new = fmaxf(m_run, mx);
    float alpha = __expf(m_run - m_new);
    float ssum = 0.f;
#pragma unroll
    for (int r = 0; r < 8; ++r) {
      S[0][r] = __expf(S[0][r] - m_new); ssum += S[0][r];
      S[1][r] = __expf(S[1][r] - m_new); ssum += S[1][r];
    }
    ssum += __shfl_xor(ssum, 16);
    l_run = l_run * alpha + ssum;
    m_run = m_new;
#pragma unroll
    for (int r = 0; r < 8; ++r) { O0[r] *= alpha; O1[r] *= alpha; }

    // assemble P^T B-frag (32 keys x 16 queries) from C-frags + lane-16 exchange
    unsigned pa[4], pb[4], pax[4], pbx[4];
#pragma unroll
    for (int j = 0; j < 4; ++j) {
      pa[j] = pk2(S[0][2 * j], S[0][2 * j + 1]);
      pb[j] = pk2(S[1][2 * j], S[1][2 * j + 1]);
    }
#pragma unroll
    for (int j = 0; j < 4; ++j) { pax[j] = __shfl_xor(pa[j], 16); pbx[j] = __shfl_xor(pb[j], 16); }
    union { unsigned u[8]; v16h h; } pf;
#pragma unroll
    for (int j = 0; j < 4; ++j) {
      pf.u[j]     = hi ? pbx[j] : pa[j];   // low lanes: keys 0-15, high lanes: keys 16-31
      pf.u[4 + j] = hi ? pb[j]  : pax[j];
    }
    int k0 = g0 + (hi ? 16 : 0), k1 = g0 + (hi ? 24 : 8);
    v16h av0 = afrag_load(v0row, k0, k1);  // V^T A-frags (dh rows, key K)
    v16h av1 = afrag_load(v1row, k0, k1);
    O0 = WMMA_F16(av0, pf.h, O0);
    O1 = WMMA_F16(av1, pf.h, O1);
  }
  float inv = 1.f / l_run;
  half_t* orow = attout + ((size_t)b * HWQ + T0 + l16) * INNER_ + hd * 32 + hi * 8;
#pragma unroll
  for (int j = 0; j < 4; ++j) {
    *(unsigned*)(orow + 2 * j)      = pk2(O0[2 * j] * inv, O0[2 * j + 1] * inv);
    *(unsigned*)(orow + 16 + 2 * j) = pk2(O1[2 * j] * inv, O1[2 * j + 1] * inv);
  }
}

// ================= kernel 3: out-proj + skip + LN + MLP + LN =================
__global__ __launch_bounds__(64) void epi_kernel(
    const half_t* __restrict__ attout, const half_t* __restrict__ Wph, const float* __restrict__ bp,
    const float* __restrict__ skip,
    const float* __restrict__ ln1g, const float* __restrict__ ln1b,
    const half_t* __restrict__ W1h, const float* __restrict__ b1,
    const half_t* __restrict__ W2h, const float* __restrict__ b2,
    const float* __restrict__ ln2g, const float* __restrict__ ln2b,
    float* __restrict__ out)
{
  __shared__ alignas(16) float  zb_s[2][16 * 128];
  __shared__ alignas(16) half_t at_s[2][16 * 128];
  __shared__ alignas(16) half_t ht_s[2][16 * 256];
  int w = threadIdx.x >> 5;
  int lane = threadIdx.x & 31, l16 = lane & 15, hi = lane >> 4;
  int wave = blockIdx.x * 2 + w;           // 128 waves: (b, qtile)
  int b = wave >> 6, qt = wave & 63;
  int T0 = qt * 16;
  float*  zb = zb_s[w];
  half_t* at = at_s[w];
  half_t* ht = ht_s[w];
  const half_t* arow = attout + ((size_t)b * HWQ + T0) * INNER_;

  // z = attout @ Wp^T + bp + skip
#pragma unroll
  for (int nt = 0; nt < 8; ++nt) {
    int o = nt * 16 + l16;
    v8f acc;
#pragma unroll
    for (int r = 0; r < 8; ++r) acc[r] = bp[o];
#pragma unroll
    for (int kc = 0; kc < 4; ++kc) {
      v16h a  = afrag_load(arow + l16 * 128, kc * 32 + (hi ? 16 : 0), kc * 32 + (hi ? 24 : 8));
      v16h bf = bfrag_wh(Wph, 128, o, kc * 32 + hi * 16);
      acc = WMMA_F16(a, bf, acc);
    }
    const float* srow = skip + ((size_t)b * DIMC + o) * HWQ + T0 + hi * 8;
#pragma unroll
    for (int r = 0; r < 8; ++r) acc[r] += srow[r];
#pragma unroll
    for (int r = 0; r < 8; ++r) zb[(r + hi * 8) * 128 + o] = acc[r];
  }
  LDS_FENCE();

  // LN1 -> at (f16, for MLP A) and zb (f32 residual)
  {
    float s1 = 0.f, s2 = 0.f;
    int cbase = hi * 64;
#pragma unroll 8
    for (int c = 0; c < 64; ++c) { float v = zb[l16 * 128 + cbase + c]; s1 += v; s2 += v * v; }
    s1 += __shfl_xor(s1, 16);
    s2 += __shfl_xor(s2, 16);
    float mean = s1 * (1.f / 128.f);
    float rstd = rsqrtf(s2 * (1.f / 128.f) - mean * mean + 1e-5f);
#pragma unroll 8
    for (int c = 0; c < 64; ++c) {
      int cc = cbase + c;
      float v = (zb[l16 * 128 + cc] - mean) * rstd * ln1g[cc] + ln1b[cc];
      at[l16 * 128 + cc] = (_Float16)v;
      zb[l16 * 128 + cc] = v;
    }
  }
  LDS_FENCE();

  // h = gelu(z @ W1^T + b1)  (16 x 256)
#pragma unroll
  for (int nt = 0; nt < 16; ++nt) {
    int o = nt * 16 + l16;
    v8f acc;
#pragma unroll
    for (int r = 0; r < 8; ++r) acc[r] = b1[o];
#pragma unroll
    for (int kc = 0; kc < 4; ++kc) {
      v16h a  = afrag_load(at + l16 * 128, kc * 32 + (hi ? 16 : 0), kc * 32 + (hi ? 24 : 8));
      v16h bf = bfrag_wh(W1h, 128, o, kc * 32 + hi * 16);
      acc = WMMA_F16(a, bf, acc);
    }
#pragma unroll
    for (int r = 0; r < 8; ++r) {
      float xg = acc[r];
      float g  = 0.5f * xg * (1.f + erff(xg * 0.70710678118f));
      ht[(r + hi * 8) * 256 + o] = (_Float16)g;
    }
  }
  LDS_FENCE();

  // z2 = h @ W2^T + b2 + z (residual), back into zb
#pragma unroll
  for (int nt = 0; nt < 8; ++nt) {
    int o = nt * 16 + l16;
    v8f acc;
#pragma unroll
    for (int r = 0; r < 8; ++r) acc[r] = b2[o];
#pragma unroll
    for (int kc = 0; kc < 8; ++kc) {
      v16h a  = afrag_load(ht + l16 * 256, kc * 32 + (hi ? 16 : 0), kc * 32 + (hi ? 24 : 8));
      v16h bf = bfrag_wh(W2h, 256, o, kc * 32 + hi * 16);
      acc = WMMA_F16(a, bf, acc);
    }
#pragma unroll
    for (int r = 0; r < 8; ++r) {
      float v = acc[r] + zb[(r + hi * 8) * 128 + o];
      zb[(r + hi * 8) * 128 + o] = v;
    }
  }
  LDS_FENCE();

  // LN2 + scatter to (B, 128, 32, 32)
  {
    float s1 = 0.f, s2 = 0.f;
    int cbase = hi * 64;
#pragma unroll 8
    for (int c = 0; c < 64; ++c) { float v = zb[l16 * 128 + cbase + c]; s1 += v; s2 += v * v; }
    s1 += __shfl_xor(s1, 16);
    s2 += __shfl_xor(s2, 16);
    float mean = s1 * (1.f / 128.f);
    float rstd = rsqrtf(s2 * (1.f / 128.f) - mean * mean + 1e-5f);
#pragma unroll 8
    for (int c = 0; c < 64; ++c) {
      int cc = cbase + c;
      float v = (zb[l16 * 128 + cc] - mean) * rstd * ln2g[cc] + ln2b[cc];
      out[((size_t)b * DIMC + cc) * HWQ + T0 + l16] = v;
    }
  }
}

// ================= host launch =================
extern "C" void kernel_launch(void* const* d_in, const int* in_sizes, int n_in,
                              void* d_out, int out_size, void* d_ws, size_t ws_size,
                              hipStream_t stream) {
  (void)in_sizes; (void)n_in; (void)out_size; (void)ws_size;
  const float* q     = (const float*)d_in[0];
  const float* k     = (const float*)d_in[1];
  const float* v     = (const float*)d_in[2];
  const float* skip  = (const float*)d_in[3];
  const float* lnq_g = (const float*)d_in[4];
  const float* lnq_b = (const float*)d_in[5];
  const float* Wq    = (const float*)d_in[6];
  const float* bq    = (const float*)d_in[7];
  const float* lnk_g = (const float*)d_in[8];
  const float* lnk_b = (const float*)d_in[9];
  const float* Wk    = (const float*)d_in[10];
  const float* bk    = (const float*)d_in[11];
  const float* lnv_g = (const float*)d_in[12];
  const float* lnv_b = (const float*)d_in[13];
  const float* Wv    = (const float*)d_in[14];
  const float* bv    = (const float*)d_in[15];
  const float* Wp    = (const float*)d_in[16];
  const float* bp    = (const float*)d_in[17];
  const float* ln1g  = (const float*)d_in[18];
  const float* ln1b  = (const float*)d_in[19];
  const float* W1    = (const float*)d_in[20];
  const float* b1    = (const float*)d_in[21];
  const float* W2    = (const float*)d_in[22];
  const float* b2    = (const float*)d_in[23];
  const float* ln2g  = (const float*)d_in[24];
  const float* ln2b  = (const float*)d_in[25];

  half_t* ws     = (half_t*)d_ws;
  half_t* qproj  = ws;                      // 2*6*1024*128   = 1,572,864 halfs
  half_t* kproj  = ws + 1572864;            // 2*10080*128    = 2,580,480 halfs
  half_t* vt     = ws + 4153344;            // 2*128*10080    = 2,580,480 halfs
  half_t* attout = ws + 6733824;            // 2*1024*128     =   262,144 halfs
  half_t* whq    = ws + 6995968;            // 128*128
  half_t* whk    = whq + 16384;
  half_t* whv    = whk + 16384;
  half_t* whp    = whv + 16384;
  half_t* wh1    = whp + 16384;             // 256*128
  half_t* wh2    = wh1 + 32768;             // 128*256

  // one-time weight converts (f32 -> f16), stream-ordered before consumers
  cvt_kernel<<<64, 256, 0, stream>>>(Wq, whq, 16384);
  cvt_kernel<<<64, 256, 0, stream>>>(Wk, whk, 16384);
  cvt_kernel<<<64, 256, 0, stream>>>(Wv, whv, 16384);
  cvt_kernel<<<64, 256, 0, stream>>>(Wp, whp, 16384);
  cvt_kernel<<<128, 256, 0, stream>>>(W1, wh1, 32768);
  cvt_kernel<<<128, 256, 0, stream>>>(W2, wh2, 32768);

  const float scale = 0.17677669529663687f; // DIM_HEAD^-0.5 folded into q-proj

  proj_kernel<<<192, 128, 0, stream>>>(q, lnq_g, lnq_b, whq, bq, qproj, HWQ, scale, 0);
  proj_kernel<<<315, 128, 0, stream>>>(k, lnk_g, lnk_b, whk, bk, kproj, HWK, 1.f, 0);
  proj_kernel<<<315, 128, 0, stream>>>(v, lnv_g, lnv_b, whv, bv, vt,    HWK, 1.f, 1);
  attn_kernel<<<128, 128, 0, stream>>>(qproj, kproj, vt, attout);
  epi_kernel<<<64, 64, 0, stream>>>(attout, whp, bp, skip, ln1g, ln1b, wh1, b1,
                                    wh2, b2, ln2g, ln2b, (float*)d_out);
}
